// CGCNN_21827023798823
// MI455X (gfx1250) — compile-verified, matrix operand order checked by reading, not run on hardware
//
#include <hip/hip_runtime.h>
#include <hip/hip_bf16.h>

#define NN    20000
#define NE    320000
#define NG    64
#define NC    256      // atom feats
#define ND    128      // bond feats
#define NL    3
#define KEDGE 640      // 2C+D
#define KBE   20       // KEDGE/32
#define KLIN  1024     // C*(L+1)
#define KBL   32       // KLIN/32
#define MT    32       // edges per tile
#define TPB   4        // tiles per workgroup (double-buffered)
#define ROWPAD 648     // 640 + 8 halves pad -> conflict-free LDS

typedef __attribute__((ext_vector_type(16))) __bf16 v16bf;
typedef __attribute__((ext_vector_type(8)))  float  v8f;

union Frag {
  v16bf v;
  uint4 q[2];
};

__device__ __forceinline__ unsigned short f2bf(float f) {
  union { float f; unsigned u; } x; x.f = f;
  unsigned r = x.u + 0x7fffu + ((x.u >> 16) & 1u);   // RNE
  return (unsigned short)(r >> 16);
}
__device__ __forceinline__ unsigned pk2(float a, float b) {
  return (unsigned)f2bf(a) | ((unsigned)f2bf(b) << 16);
}
__device__ __forceinline__ v8f wmma_bf16(v16bf a, v16bf b, v8f c) {
  return __builtin_amdgcn_wmma_f32_16x16x32_bf16(false, a, false, b, (short)0, c,
                                                 false, false);
}
__device__ __forceinline__ void atomAdd(float* p, float v) {
  unsafeAtomicAdd(p, v);   // global_atomic_add_f32 path
}
// Async DMA: 16B per lane, global -> LDS, tracked by ASYNCcnt (CDNA5).
__device__ __forceinline__ void async_g2l_b128(unsigned lds_byte_addr, const void* gaddr) {
  asm volatile("global_load_async_to_lds_b128 %0, %1, off"
               :: "v"(lds_byte_addr), "v"(gaddr) : "memory");
}
__device__ __forceinline__ void wait_async0() {
  asm volatile("s_wait_asynccnt 0x0" ::: "memory");
}

// ---- h0 init: fp32 copy + bf16 copy + concat section 0 ----
__global__ void k_init(const float* __restrict__ x, float* __restrict__ h,
                       unsigned short* __restrict__ hbf,
                       unsigned short* __restrict__ hcat) {
  int i = blockIdx.x * blockDim.x + threadIdx.x;
  if (i >= NN * NC) return;
  float v = x[i];
  h[i] = v;
  unsigned short b = f2bf(v);
  hbf[i] = b;
  int r = i >> 8, c = i & 255;
  hcat[(size_t)r * KLIN + c] = b;
}

// ---- pre-convert edge_attr to bf16 (read once, reused by all 3 layers) ----
__global__ void k_eabf(const float* __restrict__ ea, unsigned* __restrict__ eabf2) {
  int i = blockIdx.x * blockDim.x + threadIdx.x;   // pair index
  if (i >= NE * ND / 2) return;
  const float2 f = reinterpret_cast<const float2*>(ea)[i];
  eabf2[i] = pk2(f.x, f.y);
}

// ---- pack Wf/Ws into WMMA B-fragment layout ----
// packed idx = ((((l*2+mat)*16 + t)*KBE + kb)*32 + lane)*16 + i
// k = kb*32 + (lane>>4)*8 + (i&8 ? 16 : 0) + (i&7);  n = t*16 + (lane&15)
__global__ void k_packW(const float* __restrict__ Wf, const float* __restrict__ Ws,
                        unsigned short* __restrict__ pW) {
  int idx = blockIdx.x * blockDim.x + threadIdx.x;
  if (idx >= NL * 2 * 16 * KBE * 32 * 16) return;
  int i = idx & 15;
  int lane = (idx >> 4) & 31;
  int r = idx >> 9;
  int kb = r % KBE; r /= KBE;
  int t = r & 15; r >>= 4;
  int mat = r & 1;
  int l = r >> 1;
  int k = kb * 32 + ((lane >> 4) << 3) + ((i & 8) ? 16 : 0) + (i & 7);
  int n = t * 16 + (lane & 15);
  const float* W = mat ? Ws : Wf;
  pW[idx] = f2bf(W[((size_t)l * KEDGE + k) * NC + n]);
}

// ---- pack Wlin: idx = ((t*KBL + kb)*32 + lane)*16 + i ----
__global__ void k_packWlin(const float* __restrict__ Wlin,
                           unsigned short* __restrict__ pL) {
  int idx = blockIdx.x * blockDim.x + threadIdx.x;
  if (idx >= 16 * KBL * 32 * 16) return;
  int i = idx & 15;
  int lane = (idx >> 4) & 31;
  int kb = (idx >> 9) & 31;
  int t = idx >> 14;
  int k = kb * 32 + ((lane >> 4) << 3) + ((i & 8) ? 16 : 0) + (i & 7);
  int n = t * 16 + (lane & 15);
  pL[idx] = f2bf(Wlin[(size_t)k * NC + n]);
}

// ---- edge kernel: double-buffered async gather -> dual WMMA GEMM -> gate -> scatter ----
__global__ __launch_bounds__(256) void k_edge(
    int l,
    const unsigned short* __restrict__ hbf,
    const int* __restrict__ ei,
    const unsigned short* __restrict__ eabf,
    const float* __restrict__ bfv,
    const float* __restrict__ bsv,
    const unsigned short* __restrict__ pW,
    float* __restrict__ agg) {
  extern __shared__ unsigned short zl[];   // 2 * MT * ROWPAD halves
  const int tid = threadIdx.x;
  const int lane = tid & 31;
  const int w = tid >> 5;
  const int* srcp = ei;
  const int* dstp = ei + NE;
  // Runtime LDS byte base of the dynamic-shared buffer (low 32 bits of the
  // generic address == LDS offset). Computed once; no static pointer tables.
  const unsigned zbase = (unsigned)(uintptr_t)(void*)&zl[0];

  const int gr = tid >> 3, gp = tid & 7;   // 8 threads per edge row in gather

  // Issue async gather of tile t into buffer bsel: z = [h[dst] | h[src] | ea_bf16]
  auto issue = [&](int t, int bsel) {
    const int e = t * MT + gr;
    const int dn = dstp[e], sn = srcp[e];
    const unsigned short* gd = hbf + (size_t)dn * NC + gp * 32;
    const unsigned short* gs = hbf + (size_t)sn * NC + gp * 32;
    const unsigned short* ge = eabf + (size_t)e * ND + gp * 16;
    const unsigned base = zbase + (unsigned)(bsel * MT * ROWPAD + gr * ROWPAD) * 2u;
    const unsigned ld = base + gp * 64;          // bytes
    const unsigned ls = base + 512 + gp * 64;
    const unsigned le = base + 1024 + gp * 32;
#pragma unroll
    for (int j = 0; j < 4; ++j) async_g2l_b128(ld + j * 16, gd + j * 8);
#pragma unroll
    for (int j = 0; j < 4; ++j) async_g2l_b128(ls + j * 16, gs + j * 8);
#pragma unroll
    for (int j = 0; j < 2; ++j) async_g2l_b128(le + j * 16, ge + j * 8);
  };

  const int msub = w & 1;     // 2 M-subtiles of 16 edges
  const int nq = w >> 1;      // 4 N-quarters of 64 cols
  const size_t layerF = (size_t)(l * 2 + 0) * 16;
  const size_t layerS = (size_t)(l * 2 + 1) * 16;
  const int tile0 = blockIdx.x * TPB;

  issue(tile0, 0);

  for (int it = 0; it < TPB; ++it) {
    const unsigned short* cb = zl + (it & 1) * (MT * ROWPAD);
    wait_async0();        // my async writes for cb have landed in LDS
    __syncthreads();      // everyone done with prior use of the other buffer too
    if (it + 1 < TPB) issue(tile0 + it + 1, (it + 1) & 1);

    const int ebase = (tile0 + it) * MT;
    v8f accf[4] = {};
    v8f accs[4] = {};
    const unsigned short* arow =
        &cb[(msub * 16 + (lane & 15)) * ROWPAD + ((lane >> 4) << 3)];

    for (int kb = 0; kb < KBE; ++kb) {
      Frag A;
      const uint4* ap = reinterpret_cast<const uint4*>(arow + kb * 32);
      A.q[0] = ap[0];   // K = kbase .. kbase+7
      A.q[1] = ap[2];   // K = kbase+16 .. kbase+23
#pragma unroll
      for (int j = 0; j < 4; ++j) {
        const size_t t = nq * 4 + j;
        Frag Bf, Bs;
        const uint4* bpf = reinterpret_cast<const uint4*>(
            pW + (((layerF + t) * KBE + kb) * 32 + lane) * 16);
        Bf.q[0] = bpf[0]; Bf.q[1] = bpf[1];
        accf[j] = wmma_bf16(A.v, Bf.v, accf[j]);
        const uint4* bps = reinterpret_cast<const uint4*>(
            pW + (((layerS + t) * KBE + kb) * 32 + lane) * 16);
        Bs.q[0] = bps[0]; Bs.q[1] = bps[1];
        accs[j] = wmma_bf16(A.v, Bs.v, accs[j]);
      }
    }

    int dnode[8];
#pragma unroll
    for (int i = 0; i < 8; ++i) {
      const int e = ebase + msub * 16 + i + ((lane & 16) ? 8 : 0);
      dnode[i] = dstp[e];
    }
#pragma unroll
    for (int j = 0; j < 4; ++j) {
      const int col = nq * 64 + j * 16 + (lane & 15);
      const float bb = bfv[l * NC + col];
      const float sb = bsv[l * NC + col];
#pragma unroll
      for (int i = 0; i < 8; ++i) {
        float f = accf[j][i] + bb;
        float s = accs[j][i] + sb;
        float sg = 1.f / (1.f + __expf(-f));
        float sp = fmaxf(s, 0.f) + log1pf(__expf(-fabsf(s)));   // stable softplus
        atomAdd(agg + (size_t)dnode[i] * NC + col, sg * sp);
      }
    }
  }
}

// ---- node update: h = elu(h + agg); refresh bf16 + concat section l+1 ----
__global__ void k_node(int l, float* __restrict__ h, const float* __restrict__ agg,
                       unsigned short* __restrict__ hbf,
                       unsigned short* __restrict__ hcat) {
  int i = blockIdx.x * blockDim.x + threadIdx.x;
  if (i >= NN * NC) return;
  float v = h[i] + agg[i];
  v = v > 0.f ? v : expm1f(v);
  h[i] = v;
  unsigned short b = f2bf(v);
  hbf[i] = b;
  int r = i >> 8, c = i & 255;
  hcat[(size_t)r * KLIN + (size_t)(l + 1) * NC + c] = b;
}

// ---- final linear + per-graph reduction ----
__global__ __launch_bounds__(256) void k_final(
    const unsigned short* __restrict__ hcat,
    const unsigned short* __restrict__ pL,
    const float* __restrict__ blin,
    const int* __restrict__ batch,
    float* __restrict__ dout) {
  const int tid = threadIdx.x, lane = tid & 31, w = tid >> 5;
  const int mbase = blockIdx.x * 64;
  const int msub = w & 3, nhalf = w >> 2;
  const int arow = mbase + msub * 16 + (lane & 15);
  const bool va = arow < NN;
  const unsigned short* ab =
      hcat + (size_t)arow * KLIN + ((lane >> 4) << 3);
  v8f acc[8] = {};
  for (int kb = 0; kb < KBL; ++kb) {
    Frag A;
    if (va) {
      const uint4* ap = reinterpret_cast<const uint4*>(ab + kb * 32);
      A.q[0] = ap[0];
      A.q[1] = ap[2];
    } else {
      A.q[0] = make_uint4(0u, 0u, 0u, 0u);
      A.q[1] = make_uint4(0u, 0u, 0u, 0u);
    }
#pragma unroll
    for (int j = 0; j < 8; ++j) {
      const size_t t = nhalf * 8 + j;
      Frag B;
      const uint4* bp = reinterpret_cast<const uint4*>(
          pL + ((t * KBL + kb) * 32 + lane) * 16);
      B.q[0] = bp[0]; B.q[1] = bp[1];
      acc[j] = wmma_bf16(A.v, B.v, acc[j]);
    }
  }
  float* outg = dout;            // [NG, NC]
  float* nrep = dout + NG * NC;  // [NN, NC]
#pragma unroll
  for (int j = 0; j < 8; ++j) {
    const int col = nhalf * 128 + j * 16 + (lane & 15);
    const float bl = blin[col];
#pragma unroll
    for (int i = 0; i < 8; ++i) {
      const int row = mbase + msub * 16 + i + ((lane & 16) ? 8 : 0);
      if (row < NN) {
        const float v = acc[j][i] + bl;
        nrep[(size_t)row * NC + col] = v;
        atomAdd(outg + (size_t)batch[row] * NC + col, v);
      }
    }
  }
}

extern "C" void kernel_launch(void* const* d_in, const int* in_sizes, int n_in,
                              void* d_out, int out_size, void* d_ws, size_t ws_size,
                              hipStream_t stream) {
  const float* x     = (const float*)d_in[0];
  const int*   ei    = (const int*)d_in[1];
  const float* ea    = (const float*)d_in[2];
  const int*   batch = (const int*)d_in[3];
  const float* Wf    = (const float*)d_in[4];
  const float* bfv   = (const float*)d_in[5];
  const float* Ws    = (const float*)d_in[6];
  const float* bsv   = (const float*)d_in[7];
  const float* Wlin  = (const float*)d_in[8];
  const float* blin  = (const float*)d_in[9];
  float* out = (float*)d_out;
  (void)in_sizes; (void)n_in; (void)out_size; (void)ws_size;

  char* p = (char*)d_ws;
  auto take = [&](size_t bytes) -> char* {
    char* q = p;
    p += (bytes + 255) & ~(size_t)255;
    return q;
  };
  float*          h_cur = (float*)take((size_t)NN * NC * 4);
  float*          agg   = (float*)take((size_t)NN * NC * 4);
  unsigned short* hbf   = (unsigned short*)take((size_t)NN * NC * 2);
  unsigned short* hcat  = (unsigned short*)take((size_t)NN * KLIN * 2);
  unsigned short* eabf  = (unsigned short*)take((size_t)NE * ND * 2);
  unsigned short* pW    = (unsigned short*)take((size_t)NL * 2 * 16 * KBE * 32 * 16 * 2);
  unsigned short* pL    = (unsigned short*)take((size_t)16 * KBL * 32 * 16 * 2);

  k_init<<<(NN * NC + 255) / 256, 256, 0, stream>>>(x, h_cur, hbf, hcat);
  k_eabf<<<(NE * ND / 2 + 255) / 256, 256, 0, stream>>>(ea, (unsigned*)eabf);
  k_packW<<<(NL * 2 * 16 * KBE * 32 * 16) / 256, 256, 0, stream>>>(Wf, Ws, pW);
  k_packWlin<<<(16 * KBL * 32 * 16) / 256, 256, 0, stream>>>(Wlin, pL);

  for (int l = 0; l < NL; ++l) {
    (void)hipMemsetAsync(agg, 0, (size_t)NN * NC * 4, stream);
    k_edge<<<NE / (MT * TPB), 256, 2 * MT * ROWPAD * 2, stream>>>(
        l, hbf, ei, eabf, bfv, bsv, pW, agg);
    k_node<<<(NN * NC + 255) / 256, 256, 0, stream>>>(l, h_cur, agg, hbf, hcat);
  }

  (void)hipMemsetAsync(out, 0, (size_t)NG * NC * 4, stream);
  k_final<<<(NN + 63) / 64, 256, 0, stream>>>(hcat, pL, blin, batch, out);
}
